// GNNClassifier_48979807044289
// MI455X (gfx1250) — compile-verified, matrix operand order checked by reading, not run on hardware
//
#include <hip/hip_runtime.h>

typedef __attribute__((ext_vector_type(16))) __bf16 v16bf;
typedef __attribute__((ext_vector_type(8)))  float  v8f;

#define BB 16384
#define HH 768
#define NN 12
#define LL 2
#define CC 4
#define DD 64
#define SB 16              // samples per workgroup
#define ROWS (SB*NN)       // 192 rows of h per workgroup
#define HFS 65             // padded f32 row stride (dwords)
#define HBS 66             // padded bf16 row stride (elements, even -> dword aligned)
#define MMS 65
#define AAS 13

static __device__ __forceinline__ __bf16 f2bf(float f) {
    return static_cast<__bf16>(f);   // native v_cvt bf16 on gfx1250
}

__global__ __launch_bounds__(128)
void gnn_fused_kernel(const float* __restrict__ x,
                      const float* __restrict__ mp_w,
                      const float* __restrict__ mp_b,
                      const float* __restrict__ attn_w,
                      const float* __restrict__ attn_b,
                      const float* __restrict__ ln_g,
                      const float* __restrict__ ln_b,
                      const float* __restrict__ cls_w,
                      const float* __restrict__ cls_b,
                      float* __restrict__ out)
{
    __shared__ float  hf[ROWS * HFS];       // h tile, f32 (residual / LN precision)
    __shared__ __bf16 hb[ROWS * HBS];       // h tile, bf16 (WMMA A operand source)
    __shared__ float  mm[ROWS * MMS];       // m = h @ W^T
    __shared__ float  aa[ROWS * AAS];       // softmax rows
    __shared__ float  eis[ROWS], ejs[ROWS];
    __shared__ float  w1s[DD], w2s[DD], mpb[DD], lng[DD], lnbs[DD];

    const int tid  = threadIdx.x;
    const int lane = tid & 31;
    const int wave = tid >> 5;
    const int b0   = blockIdx.x * SB;

    // ---- stage x tile (ROWS x 64 f32, contiguous in global) into LDS ----
    const float4* xsrc = (const float4*)(x + (size_t)b0 * HH);
    for (int i = tid; i < ROWS * DD / 4; i += 128) {
        float4 v = xsrc[i];
        int e = i * 4;
        int r = e / DD, c = e % DD;
        float* dst = &hf[r * HFS + c];
        dst[0] = v.x; dst[1] = v.y; dst[2] = v.z; dst[3] = v.w;
        __bf16* db = &hb[r * HBS + c];
        db[0] = f2bf(v.x); db[1] = f2bf(v.y); db[2] = f2bf(v.z); db[3] = f2bf(v.w);
    }
    if (tid < CC) __builtin_prefetch(cls_w + tid * HH, 0, 1);  // global_prefetch
    __syncthreads();

    for (int l = 0; l < LL; ++l) {
        const float* Wl = mp_w + l * DD * DD;
        if (tid < DD) {
            w1s[tid]  = attn_w[l * 2 * DD + tid];
            w2s[tid]  = attn_w[l * 2 * DD + DD + tid];
            mpb[tid]  = mp_b[l * DD + tid];
            lng[tid]  = ln_g[l * DD + tid];
            lnbs[tid] = ln_b[l * DD + tid];
        }
        __syncthreads();

        // ---- attention scores ei = h.w1, ej = h.w2 ----
        for (int r = tid; r < ROWS; r += 128) {
            const float* hr = &hf[r * HFS];
            float s1 = 0.f, s2 = 0.f;
            #pragma unroll 8
            for (int d = 0; d < DD; ++d) { s1 += hr[d] * w1s[d]; s2 += hr[d] * w2s[d]; }
            eis[r] = s1; ejs[r] = s2;
        }
        __syncthreads();

        // ---- softmax over j of (ei + ej + b), per (sample,i) row ----
        const float ab = attn_b[l];
        for (int r = tid; r < ROWS; r += 128) {
            int   s    = r / NN;
            float base = eis[r] + ab;
            float ev[NN];
            float mx = -3.0e38f;
            #pragma unroll
            for (int j = 0; j < NN; ++j) { float e = base + ejs[s * NN + j]; ev[j] = e; mx = fmaxf(mx, e); }
            float sum = 0.f;
            #pragma unroll
            for (int j = 0; j < NN; ++j) { float e = __expf(ev[j] - mx); ev[j] = e; sum += e; }
            float inv = 1.f / sum;
            #pragma unroll
            for (int j = 0; j < NN; ++j) aa[r * AAS + j] = ev[j] * inv;
        }

        // ---- B fragments (mp_w as bf16) in registers; B[k][n] = W[n][k] ----
        // 16-bit operand VGPR layout: K = kt*32 + 16*(v/4) + 8*half + 2*(v%4) + p
        v16bf bB[4][2];
        {
            const int n    = lane & 15;
            const int half = lane >> 4;
            #pragma unroll
            for (int nt = 0; nt < 4; ++nt) {
                const float* wcol = Wl + (nt * 16 + n) * DD;
                #pragma unroll
                for (int kt = 0; kt < 2; ++kt) {
                    v16bf f;
                    #pragma unroll
                    for (int v = 0; v < 8; ++v) {
                        int k = kt * 32 + 16 * (v >> 2) + 8 * half + 2 * (v & 3);
                        f[2 * v]     = f2bf(wcol[k]);
                        f[2 * v + 1] = f2bf(wcol[k + 1]);
                    }
                    bB[nt][kt] = f;
                }
            }
        }
        __syncthreads();

        // ---- m = h @ W^T via v_wmma_f32_16x16x32_bf16 (3 M-tiles per wave) ----
        // A fragments come straight from the bf16 LDS copy: pure ds_load_b32.
        {
            const int mrow = lane & 15;
            const int half = lane >> 4;
            for (int mt = wave * 3; mt < wave * 3 + 3; ++mt) {
                int row = mt * 16 + mrow;
                v16bf aF[2];
                #pragma unroll
                for (int kt = 0; kt < 2; ++kt) {
                    v16bf f;
                    #pragma unroll
                    for (int v = 0; v < 8; ++v) {
                        int k = kt * 32 + 16 * (v >> 2) + 8 * half + 2 * (v & 3);
                        const __bf16* hp = &hb[row * HBS + k];
                        f[2 * v]     = hp[0];
                        f[2 * v + 1] = hp[1];
                    }
                    aF[kt] = f;
                }
                #pragma unroll
                for (int nt = 0; nt < 4; ++nt) {
                    v8f acc = {};
                    acc = __builtin_amdgcn_wmma_f32_16x16x32_bf16(
                            false, aF[0], false, bB[nt][0], (short)0, acc, false, false);
                    acc = __builtin_amdgcn_wmma_f32_16x16x32_bf16(
                            false, aF[1], false, bB[nt][1], (short)0, acc, false, false);
                    // D layout: M = r8 + 8*half, N = lane%16 (+16*nt)
                    #pragma unroll
                    for (int r8 = 0; r8 < 8; ++r8)
                        mm[(mt * 16 + half * 8 + r8) * MMS + nt * 16 + mrow] = acc[r8];
                }
            }
        }
        __syncthreads();

        // ---- agg = a @ m (+mp_b folded: softmax row sums to 1), residual, exact GELU ----
        for (int r = tid; r < ROWS; r += 128) {
            int s = r / NN;
            float ar[NN];
            #pragma unroll
            for (int j = 0; j < NN; ++j) ar[j] = aa[r * AAS + j];
            float* hr = &hf[r * HFS];
            for (int d = 0; d < DD; ++d) {
                float acc = mpb[d];
                #pragma unroll
                for (int j = 0; j < NN; ++j) acc += ar[j] * mm[(s * NN + j) * MMS + d];
                float xv = hr[d] + acc;
                hr[d] = 0.5f * xv * (1.0f + erff(xv * 0.70710678118654752f));
            }
        }
        __syncthreads();

        // ---- LayerNorm over d; refresh both f32 and bf16 copies ----
        for (int r = tid; r < ROWS; r += 128) {
            float* hr = &hf[r * HFS];
            float mu = 0.f;
            #pragma unroll 8
            for (int d = 0; d < DD; ++d) mu += hr[d];
            mu *= (1.0f / DD);
            float var = 0.f;
            #pragma unroll 8
            for (int d = 0; d < DD; ++d) { float t = hr[d] - mu; var += t * t; }
            var *= (1.0f / DD);
            float inv = rsqrtf(var + 1e-5f);
            __bf16* hbr = &hb[r * HBS];
            #pragma unroll 4
            for (int d = 0; d < DD; ++d) {
                float v = (hr[d] - mu) * inv * lng[d] + lnbs[d];
                hr[d]  = v;
                hbr[d] = f2bf(v);
            }
        }
        __syncthreads();
    }

    // ---- classifier: per sample, 4 dots of length 768 ----
    if (tid < SB * CC) {
        int s = tid >> 2;
        int c = tid & 3;
        const float* wc = cls_w + c * HH;
        float acc = cls_b[c];
        for (int n = 0; n < NN; ++n) {
            const float* hr = &hf[(s * NN + n) * HFS];
            #pragma unroll 8
            for (int d = 0; d < DD; ++d) acc += hr[d] * wc[n * DD + d];
        }
        out[(size_t)(b0 + s) * CC + c] = acc;
    }
}

extern "C" void kernel_launch(void* const* d_in, const int* in_sizes, int n_in,
                              void* d_out, int out_size, void* d_ws, size_t ws_size,
                              hipStream_t stream) {
    const float* x      = (const float*)d_in[0];
    const float* mp_w   = (const float*)d_in[1];
    const float* mp_b   = (const float*)d_in[2];
    const float* attn_w = (const float*)d_in[3];
    const float* attn_b = (const float*)d_in[4];
    const float* ln_g   = (const float*)d_in[5];
    const float* ln_b   = (const float*)d_in[6];
    const float* cls_w  = (const float*)d_in[7];
    const float* cls_b  = (const float*)d_in[8];
    float* out = (float*)d_out;

    dim3 grid(BB / SB), block(128);
    gnn_fused_kernel<<<grid, block, 0, stream>>>(x, mp_w, mp_b, attn_w, attn_b,
                                                 ln_g, ln_b, cls_w, cls_b, out);
}
